// RNNClassifier_31456340475920
// MI455X (gfx1250) — matrix operand order repaired and checked by reading, not compiled, not run on hardware
//
#include <hip/hip_runtime.h>
#include <math.h>

// ---------------- types ----------------
typedef __attribute__((ext_vector_type(16))) __bf16 v16bf;
typedef __attribute__((ext_vector_type(8)))  __bf16 v8bf;
typedef __attribute__((ext_vector_type(8)))  float  v8f;

#define B_ROWS 4096
#define D_IN   4096
#define H_DIM  2048
#define D_OUT  1000
#define LN_EPS 1e-5f

__device__ __forceinline__ unsigned short f32_to_bf16_rne(float f) {
  unsigned int u = __float_as_uint(f);
  unsigned int r = u + 0x7FFFu + ((u >> 16) & 1u);
  return (unsigned short)(r >> 16);
}
__device__ __forceinline__ float bf16_to_f32(unsigned short s) {
  return __uint_as_float(((unsigned int)s) << 16);
}

// CDNA5 async global->LDS DMA (ASYNCcnt path, no VGPR staging)
__device__ __forceinline__ void async_ld_b128(unsigned lds_off, unsigned long long gaddr) {
  asm volatile("global_load_async_to_lds_b128 %0, %1, off"
               :: "v"(lds_off), "v"(gaddr) : "memory");
}
__device__ __forceinline__ void wait_async0() {
  asm volatile("s_wait_asynccnt 0x0" ::: "memory");
}

// ---------------- f32 -> bf16 conversion (memory-bound prologue) -------------
__global__ __launch_bounds__(256)
void convert_f32_to_bf16(const float* __restrict__ src,
                         unsigned short* __restrict__ dst, int n4) {
  // n4 = count of float4 groups (all sizes here are multiples of 4)
  int i = blockIdx.x * blockDim.x + threadIdx.x;
  int stride = gridDim.x * blockDim.x;
  for (; i < n4; i += stride) {
    float4 v = ((const float4*)src)[i];
    ushort4 o;
    o.x = f32_to_bf16_rne(v.x); o.y = f32_to_bf16_rne(v.y);
    o.z = f32_to_bf16_rne(v.z); o.w = f32_to_bf16_rne(v.w);
    ((ushort4*)dst)[i] = o;
  }
}

// ---------------- per-row LayerNorm stats over h2 (concat-folded) ------------
// combined = [h2, h2] => mean/var over 2H equal mean/var over H.
__global__ __launch_bounds__(256)
void row_stats(const unsigned short* __restrict__ h2,
               float* __restrict__ rowa, float* __restrict__ rowc) {
  __shared__ float ssum[256], ssq[256];
  const int row = blockIdx.x;
  const unsigned short* p = h2 + (size_t)row * H_DIM;
  float s = 0.f, q = 0.f;
  for (int i = threadIdx.x; i < H_DIM; i += 256) {
    float v = bf16_to_f32(p[i]);
    s += v; q += v * v;
  }
  ssum[threadIdx.x] = s; ssq[threadIdx.x] = q;
  __syncthreads();
  for (int off = 128; off > 0; off >>= 1) {
    if (threadIdx.x < off) {
      ssum[threadIdx.x] += ssum[threadIdx.x + off];
      ssq[threadIdx.x]  += ssq[threadIdx.x + off];
    }
    __syncthreads();
  }
  if (threadIdx.x == 0) {
    float mean = ssum[0] * (1.0f / H_DIM);
    float var  = ssq[0] * (1.0f / H_DIM) - mean * mean;
    float r = rsqrtf(var + LN_EPS);
    rowa[row] = r;          // scale
    rowc[row] = -mean * r;  // shift (pre gamma/beta, folded into U/e/d)
  }
}

// ---------------- fold concat+gamma into FC weights --------------------------
// U[o,k] = fcW[o,k]*g[k] + fcW[o,k+H]*g[k+H]   (bf16, [D_OUT x H])
// e[o]   = sum_k U[o,k] ; d[o] = sum_j ln_b[j]*fcW[o,j] + fc_b[o]
__global__ __launch_bounds__(256)
void prep_fc(const float* __restrict__ fcW, const float* __restrict__ fcb,
             const float* __restrict__ g, const float* __restrict__ bvec,
             unsigned short* __restrict__ U,
             float* __restrict__ evec, float* __restrict__ dvec) {
  __shared__ float se[256], sd[256];
  const int o = blockIdx.x;
  const float* wrow = fcW + (size_t)o * (2 * H_DIM);
  float esum = 0.f, dsum = 0.f;
  for (int k = threadIdx.x; k < H_DIM; k += 256) {
    float u = wrow[k] * g[k] + wrow[k + H_DIM] * g[k + H_DIM];
    U[(size_t)o * H_DIM + k] = f32_to_bf16_rne(u);
    esum += u;
    dsum += bvec[k] * wrow[k] + bvec[k + H_DIM] * wrow[k + H_DIM];
  }
  se[threadIdx.x] = esum; sd[threadIdx.x] = dsum;
  __syncthreads();
  for (int off = 128; off > 0; off >>= 1) {
    if (threadIdx.x < off) {
      se[threadIdx.x] += se[threadIdx.x + off];
      sd[threadIdx.x] += sd[threadIdx.x + off];
    }
    __syncthreads();
  }
  if (threadIdx.x == 0) { evec[o] = se[0]; dvec[o] = sd[0] + fcb[o]; }
}

// ---------------- WMMA GEMM: C = A[MxK] * W[NxK]^T, bf16 in / f32 acc --------
// Block tile 256x128xBK64, 8 waves (4x2), wave tile 64x64 = 4x4 wmma tiles.
// Double-buffered LDS fed by GLOBAL_LOAD_ASYNC_TO_LDS_B128 (ASYNCcnt).
// EPI==0: out_bf16 = tanh(acc + bias0[n] + bias1[n])
// EPI==1: out_f32  = rowa[m]*acc + rowc[m]*e[n] + d[n]  (N guarded)
template <int EPI>
__global__ __launch_bounds__(256)
void gemm_wmma_bf16(const unsigned short* __restrict__ A, int lda,
                    const unsigned short* __restrict__ W, int ldw,
                    int N, int K,
                    const float* __restrict__ bias0, const float* __restrict__ bias1,
                    const float* __restrict__ rowa, const float* __restrict__ rowc,
                    const float* __restrict__ evec, const float* __restrict__ dvec,
                    unsigned short* __restrict__ out_bf16,
                    float* __restrict__ out_f32, int ldo) {
  // padded rows: 72 ushorts = 144 bytes (16B-aligned stores, bank-spread reads)
  __shared__ __align__(16) unsigned short As[2][256][72];
  __shared__ __align__(16) unsigned short Ws[2][128][72];

  const int tid  = threadIdx.x;
  const int wid  = tid >> 5;        // wave32
  const int lane = tid & 31;
  const int half = lane >> 4;
  const int lr   = lane & 15;
  const int wrow = wid >> 1;        // 0..3 : 64-row slab
  const int wcol = wid & 1;         // 0..1 : 64-col slab
  const int m_base = blockIdx.y * 256;
  const int n_base = blockIdx.x * 128;

  // async cooperative-load mapping
  const int a_row = tid;            // A: 1 thread per row, 64 cols (8 x b128)
  const int w_row = tid >> 1;       // W: 2 threads per row, 32 cols (4 x b128)
  const int w_col = (tid & 1) * 32;
  const bool w_ok = (n_base + w_row) < N;

  const unsigned as_base = (unsigned)(uintptr_t)&As[0][0][0];
  const unsigned ws_base = (unsigned)(uintptr_t)&Ws[0][0][0];
  const unsigned a_lds_row = as_base + (unsigned)(a_row * 72) * 2u;
  const unsigned w_lds_row = ws_base + (unsigned)(w_row * 72 + w_col) * 2u;
  const unsigned long long a_grow =
      (unsigned long long)(uintptr_t)(A + (size_t)(m_base + a_row) * lda);
  const unsigned long long w_grow =
      (unsigned long long)(uintptr_t)(W + (size_t)(n_base + w_row) * ldw + w_col);

  auto issue_tile = [&](int k0, int buf) {
    const unsigned a_lds = a_lds_row + (unsigned)(buf * 256 * 72) * 2u;
    const unsigned long long a_g = a_grow + (unsigned long long)k0 * 2u;
#pragma unroll
    for (int i = 0; i < 8; ++i) async_ld_b128(a_lds + i * 16, a_g + i * 16);
    if (w_ok) {
      const unsigned w_lds = w_lds_row + (unsigned)(buf * 128 * 72) * 2u;
      const unsigned long long w_g = w_grow + (unsigned long long)k0 * 2u;
#pragma unroll
      for (int i = 0; i < 4; ++i) async_ld_b128(w_lds + i * 16, w_g + i * 16);
    }
  };

  // zero-fill W rows >= N once (never overwritten by async loads afterwards)
  if (!w_ok) {
    const uint4 z = {0, 0, 0, 0};
#pragma unroll
    for (int b = 0; b < 2; ++b)
#pragma unroll
      for (int i = 0; i < 4; ++i)
        *(uint4*)&Ws[b][w_row][w_col + i * 8] = z;
  }

  v8f acc[4][4] = {};
  union Frag { v16bf v; v8bf h[2]; };

  const int NT = K >> 6;            // K / 64
  issue_tile(0, 0);
  wait_async0();
  __syncthreads();

  for (int kb = 0; kb < NT; ++kb) {
    const int cur = kb & 1;
    if (kb + 1 < NT) issue_tile((kb + 1) << 6, cur ^ 1);  // DMA overlaps compute

#pragma unroll
    for (int ks = 0; ks < 2; ++ks) {
      // A frag: lane<16 holds row m, K {0..7,16..23}; lane>=16: K {8..15,24..31}
      Frag a[4];
#pragma unroll
      for (int mi = 0; mi < 4; ++mi) {
        const unsigned short* ap = &As[cur][wrow * 64 + mi * 16 + lr][ks * 32];
        a[mi].h[0] = *(const v8bf*)(ap + half * 8);
        a[mi].h[1] = *(const v8bf*)(ap + 16 + half * 8);
      }
      // B frag: lane<16 holds column n, K {0..7,16..23}; lane>=16: K {8..15,24..31}
      Frag b[4];
#pragma unroll
      for (int ni = 0; ni < 4; ++ni) {
        const unsigned short* bp = &Ws[cur][wcol * 64 + ni * 16 + lr][ks * 32 + half * 8];
        b[ni].h[0] = *(const v8bf*)(bp);
        b[ni].h[1] = *(const v8bf*)(bp + 16);
      }
#pragma unroll
      for (int mi = 0; mi < 4; ++mi)
#pragma unroll
        for (int ni = 0; ni < 4; ++ni)
          acc[mi][ni] = __builtin_amdgcn_wmma_f32_16x16x32_bf16(
              false, a[mi].v, false, b[ni].v, (short)0, acc[mi][ni], false, false);
    }

    wait_async0();     // next tile fully in LDS (per-wave), then block-wide sync
    __syncthreads();
  }

  // C/D layout: lane<16 -> VGPR v is (M = v,   N = lane)
  //             lane>=16-> VGPR v is (M = v+8, N = lane-16)
#pragma unroll
  for (int mi = 0; mi < 4; ++mi) {
#pragma unroll
    for (int ni = 0; ni < 4; ++ni) {
      const int n = n_base + wcol * 64 + ni * 16 + lr;
      float bsum = 0.f, ev = 0.f, dv = 0.f;
      if (EPI == 0) bsum = bias0[n] + bias1[n];
      else if (n < N) { ev = evec[n]; dv = dvec[n]; }
#pragma unroll
      for (int v = 0; v < 8; ++v) {
        const int m = m_base + wrow * 64 + mi * 16 + half * 8 + v;
        float val = acc[mi][ni][v];
        if (EPI == 0) {
          out_bf16[(size_t)m * ldo + n] = f32_to_bf16_rne(tanhf(val + bsum));
        } else {
          if (n < N)
            out_f32[(size_t)m * ldo + n] = rowa[m] * val + rowc[m] * ev + dv;
        }
      }
    }
  }
}

// ---------------- launcher ----------------
extern "C" void kernel_launch(void* const* d_in, const int* in_sizes, int n_in,
                              void* d_out, int out_size, void* d_ws, size_t ws_size,
                              hipStream_t stream) {
  (void)in_sizes; (void)n_in; (void)out_size; (void)ws_size;
  const float* x     = (const float*)d_in[0];
  const float* W_ih0 = (const float*)d_in[1];
  const float* b_ih0 = (const float*)d_in[2];
  const float* b_hh0 = (const float*)d_in[3];
  const float* W_ih1 = (const float*)d_in[4];
  const float* b_ih1 = (const float*)d_in[5];
  const float* b_hh1 = (const float*)d_in[6];
  const float* ln_g  = (const float*)d_in[7];
  const float* ln_b  = (const float*)d_in[8];
  const float* fc_W  = (const float*)d_in[9];
  const float* fc_b  = (const float*)d_in[10];
  float* out = (float*)d_out;

  // workspace layout (aliased regions, peak ~72MB):
  //  regA: xb (32MB)  -> later reused for h2b (16MB)
  //  regB: w0b (16MB) -> later reused for U / rowa / rowc / e / d
  //  regC: w1b (8MB)
  //  regD: h1b (16MB)
  char* base = (char*)d_ws;
  unsigned short* xb  = (unsigned short*)(base);
  unsigned short* h2b = (unsigned short*)(base);                            // alias of xb
  unsigned short* w0b = (unsigned short*)(base + (size_t)32 * 1024 * 1024);
  unsigned short* Ub  = (unsigned short*)(base + (size_t)32 * 1024 * 1024); // alias of w0b
  float* rowa = (float*)(base + (size_t)32 * 1024 * 1024 + 4 * 1024 * 1024);
  float* rowc = rowa + B_ROWS;
  float* evec = rowc + B_ROWS;
  float* dvec = evec + D_OUT;
  unsigned short* w1b = (unsigned short*)(base + (size_t)48 * 1024 * 1024);
  unsigned short* h1b = (unsigned short*)(base + (size_t)56 * 1024 * 1024);

  // 1) fp32 -> bf16 conversions (vectorized float4 -> ushort4)
  convert_f32_to_bf16<<<2048, 256, 0, stream>>>(x,     xb,  B_ROWS * D_IN / 4);
  convert_f32_to_bf16<<<1024, 256, 0, stream>>>(W_ih0, w0b, H_DIM * D_IN / 4);
  convert_f32_to_bf16<<<1024, 256, 0, stream>>>(W_ih1, w1b, H_DIM * H_DIM / 4);

  // 2) h1 = tanh(x @ W0^T + b_ih0 + b_hh0)   [4096 x 2048]
  gemm_wmma_bf16<0><<<dim3(H_DIM / 128, B_ROWS / 256), 256, 0, stream>>>(
      xb, D_IN, w0b, D_IN, H_DIM, D_IN,
      b_ih0, b_hh0, nullptr, nullptr, nullptr, nullptr,
      h1b, nullptr, H_DIM);

  // 3) h2 = tanh(h1 @ W1^T + b_ih1 + b_hh1)  [4096 x 2048] (written over xb)
  gemm_wmma_bf16<0><<<dim3(H_DIM / 128, B_ROWS / 256), 256, 0, stream>>>(
      h1b, H_DIM, w1b, H_DIM, H_DIM, H_DIM,
      b_ih1, b_hh1, nullptr, nullptr, nullptr, nullptr,
      h2b, nullptr, H_DIM);

  // 4) per-row LN stats; 5) fold concat+gamma/beta into FC weights
  row_stats<<<B_ROWS, 256, 0, stream>>>(h2b, rowa, rowc);
  prep_fc<<<D_OUT, 256, 0, stream>>>(fc_W, fc_b, ln_g, ln_b, Ub, evec, dvec);

  // 6) out = a_m * (h2 @ U^T) + c_m * e + d   [4096 x 1000]
  gemm_wmma_bf16<1><<<dim3((D_OUT + 127) / 128, B_ROWS / 256), 256, 0, stream>>>(
      h2b, H_DIM, Ub, H_DIM, D_OUT, H_DIM,
      nullptr, nullptr, rowa, rowc, evec, dvec,
      nullptr, out, D_OUT);
}
